// GCN_43198781063543
// MI455X (gfx1250) — matrix operand order repaired and checked by reading, not compiled, not run on hardware
//
#include <hip/hip_runtime.h>
#include <hip/hip_bf16.h>
#include <math.h>

typedef __attribute__((ext_vector_type(16))) __bf16 v16bf;
typedef __attribute__((ext_vector_type(8)))  __bf16 v8bf;
typedef __attribute__((ext_vector_type(8)))  float  v8f;

#define NNODES 50000
#define DF 4096
#define DL 256
#define DID 64

__device__ __forceinline__ float lrelu(float x) { return x >= 0.f ? x : 0.01f * x; }

__device__ __forceinline__ v8f zero8() {
  v8f z;
#pragma unroll
  for (int i = 0; i < 8; ++i) z[i] = 0.f;
  return z;
}

__device__ __forceinline__ v8f wmma_bf16(v16bf a, v16bf b, v8f c) {
  // D = A x B + C, 16x16x32 bf16 -> f32
  return __builtin_amdgcn_wmma_f32_16x16x32_bf16(false, a, false, b, (short)0, c,
                                                 false, false);
}

// ---- A-matrix fragment (16x32, MxK) from row-major fp32 [rows x ld].
// Lanes 0-15: M=lane, K in {0-7,16-23}; lanes 16-31: M=lane-16, K in {8-15,24-31}.
template <bool LRELU>
__device__ __forceinline__ v16bf load_fragA_f32(const float* __restrict__ base,
                                                int ld, int row0, int k0) {
  const int lane = threadIdx.x & 31;
  const int m = row0 + (lane & 15);
  const float* p = base + (size_t)m * ld + k0 + ((lane >> 4) << 3);
  v16bf f;
#pragma unroll
  for (int i = 0; i < 8; ++i) {
    float a = p[i];
    float c = p[i + 16];
    if (LRELU) { a = lrelu(a); c = lrelu(c); }
    f[i]     = (__bf16)a;
    f[i + 8] = (__bf16)c;
  }
  return f;
}

// ---- A-matrix fragment from pre-converted bf16 row-major [rows x ld].
// Two contiguous 16-byte runs per lane -> two b128 loads, zero VALU cvt.
__device__ __forceinline__ v16bf load_fragA_bf(const __bf16* __restrict__ base,
                                               int ld, int row0, int k0) {
  const int lane = threadIdx.x & 31;
  const int m = row0 + (lane & 15);
  const __bf16* p = base + (size_t)m * ld + k0 + ((lane >> 4) << 3);
  v8bf lo = *(const v8bf*)p;
  v8bf hi = *(const v8bf*)(p + 16);
  v16bf f;
#pragma unroll
  for (int i = 0; i < 8; ++i) { f[i] = lo[i]; f[i + 8] = hi[i]; }
  return f;
}

// ---- B-matrix fragment (32x16, KxN) with B = W^T, W bf16 row-major [N x ld].
// Lanes 0-15: N=lane, K in {0-7,16-23}; lanes 16-31: N=lane-16, K in {8-15,24-31}.
// Two contiguous 16-byte runs per lane.
__device__ __forceinline__ v16bf load_fragB_bf(const __bf16* __restrict__ W,
                                               int ld, int col0, int k0) {
  const int lane = threadIdx.x & 31;
  const int n = col0 + (lane & 15);
  const __bf16* p = W + (size_t)n * ld + k0 + ((lane >> 4) << 3);
  v8bf lo = *(const v8bf*)p;
  v8bf hi = *(const v8bf*)(p + 16);
  v16bf f;
#pragma unroll
  for (int i = 0; i < 8; ++i) { f[i] = lo[i]; f[i + 8] = hi[i]; }
  return f;
}

// ---------------- fp32 -> bf16 conversion for weights ----------------------
__global__ __launch_bounds__(256) void k_f32_to_bf16(const float* __restrict__ src,
                                                     __bf16* __restrict__ dst,
                                                     int n) {
  int i = blockIdx.x * blockDim.x + threadIdx.x;
  if (i < n) dst[i] = (__bf16)src[i];
}

// ---------------- Stage 1: x0 = l2norm(features @ W_mlp^T + b_mlp) ----------
// Block: 16 rows x 256 cols, 8 waves, each wave owns a 16x32 column strip.
__global__ __launch_bounds__(256) void k_mlp_norm(const float* __restrict__ A,
                                                  const __bf16* __restrict__ W,
                                                  const float* __restrict__ bias,
                                                  __bf16* __restrict__ out) {
  const int row0 = blockIdx.x * 16;
  const int wave = threadIdx.x >> 5;
  const int lane = threadIdx.x & 31;
  const int col0 = wave * 32;
  const float* arow = A + (size_t)(row0 + (lane & 15)) * DF + ((lane >> 4) << 3);

  v8f acc0 = zero8(), acc1 = zero8();
  for (int kt = 0; kt < DF; kt += 32) {
    // Unconditional speculative prefetch of the A stream two k-tiles ahead;
    // locality 3 -> WGP-scope (pull into all cache levels). OOB prefetch past
    // the row end is speculative and silently dropped by hardware.
    __builtin_prefetch(arow + kt + 64, 0, 3);
    v16bf a  = load_fragA_f32<false>(A, DF, row0, kt);
    v16bf b0 = load_fragB_bf(W, DF, col0, kt);
    v16bf b1 = load_fragB_bf(W, DF, col0 + 16, kt);
    acc0 = wmma_bf16(a, b0, acc0);
    acc1 = wmma_bf16(a, b1, acc1);
  }

  __shared__ float tile[16][DL + 1];
  __shared__ float part[16][16];
  __shared__ float rnorm[16];

  const int mrow = (lane >> 4) * 8;
  const int ncol = lane & 15;
#pragma unroll
  for (int v = 0; v < 8; ++v) {
    tile[mrow + v][col0 + ncol]      = acc0[v] + bias[col0 + ncol];
    tile[mrow + v][col0 + 16 + ncol] = acc1[v] + bias[col0 + 16 + ncol];
  }
  __syncthreads();

  const int r   = threadIdx.x >> 4;
  const int seg = threadIdx.x & 15;
  float s = 0.f;
#pragma unroll
  for (int j = 0; j < 16; ++j) {
    float t = tile[r][seg * 16 + j];
    s += t * t;
  }
  part[r][seg] = s;
  __syncthreads();
  if (threadIdx.x < 16) {
    float tot = 0.f;
#pragma unroll
    for (int j = 0; j < 16; ++j) tot += part[threadIdx.x][j];
    rnorm[threadIdx.x] = 1.0f / fmaxf(sqrtf(tot), 1e-12f);
  }
  __syncthreads();
  const float sc = rnorm[r];
#pragma unroll
  for (int j = 0; j < 16; ++j)
    out[(size_t)(row0 + r) * DL + seg * 16 + j] =
        (__bf16)(tile[r][seg * 16 + j] * sc);
}

// ---------------- Generic NT GEMM: out[MxN] = A[MxK] @ W[NxK]^T (bf16 in) ---
template <int KDIM, int NDIM, int TPW>
__global__ void k_gemm_nt(const __bf16* __restrict__ A,
                          const __bf16* __restrict__ W, float* __restrict__ out) {
  const int row0 = blockIdx.x * 16;
  const int wave = threadIdx.x >> 5;
  const int lane = threadIdx.x & 31;
  const int col0 = wave * 16 * TPW;

  v8f acc[TPW];
#pragma unroll
  for (int t = 0; t < TPW; ++t) acc[t] = zero8();

  for (int kt = 0; kt < KDIM; kt += 32) {
    v16bf a = load_fragA_bf(A, KDIM, row0, kt);
#pragma unroll
    for (int t = 0; t < TPW; ++t) {
      v16bf b = load_fragB_bf(W, KDIM, col0 + 16 * t, kt);
      acc[t] = wmma_bf16(a, b, acc[t]);
    }
  }

  const int mrow = (lane >> 4) * 8;
  const int ncol = lane & 15;
#pragma unroll
  for (int t = 0; t < TPW; ++t)
#pragma unroll
    for (int v = 0; v < 8; ++v)
      out[(size_t)(row0 + mrow + v) * NDIM + col0 + 16 * t + ncol] = acc[t][v];
}

// ---------------- Edge scatter: agg[dst] += xw[src], one wave per edge ------
__global__ __launch_bounds__(256) void k_scatter(const float* __restrict__ xw,
                                                 const int* __restrict__ ei, int E,
                                                 int D, float* __restrict__ agg) {
  const int e = (blockIdx.x * blockDim.x + threadIdx.x) >> 5;
  const int lane = threadIdx.x & 31;
  if (e >= E) return;
  const int src = ei[e];
  const int dst = ei[E + e];
  const float* s = xw + (size_t)src * D;
  float* d = agg + (size_t)dst * D;
  for (int c = lane; c < D; c += 32) atomicAdd(&d[c], s[c]);
}

// ---- Fused layer combine:
//   out = lrelu( lrelu(agg)@Wg^T + bg + lrelu(x@Wl^T + bl) + id_emb )
template <int KDIM, typename OutT>
__global__ __launch_bounds__(128) void k_combine(
    const float* __restrict__ Hagg, const __bf16* __restrict__ X,
    const __bf16* __restrict__ Wg, const float* __restrict__ bg,
    const __bf16* __restrict__ Wl, const float* __restrict__ bl,
    const float* __restrict__ id_emb, OutT* __restrict__ out) {
  const int row0 = blockIdx.x * 16;
  const int wave = threadIdx.x >> 5;
  const int lane = threadIdx.x & 31;
  const int col0 = wave * 16;  // 4 waves cover N=64

  v8f accG = zero8(), accL = zero8();
  for (int kt = 0; kt < KDIM; kt += 32) {
    v16bf ah  = load_fragA_f32<true>(Hagg, KDIM, row0, kt);  // h = lrelu(seg_sum)
    v16bf ax  = load_fragA_bf(X, KDIM, row0, kt);
    v16bf bgf = load_fragB_bf(Wg, KDIM, col0, kt);
    v16bf blf = load_fragB_bf(Wl, KDIM, col0, kt);
    accG = wmma_bf16(ah, bgf, accG);
    accL = wmma_bf16(ax, blf, accL);
  }

  const int m0 = row0 + (lane >> 4) * 8;
  const int n  = col0 + (lane & 15);
#pragma unroll
  for (int v = 0; v < 8; ++v) {
    const int m = m0 + v;
    float xhat = lrelu(accL[v] + bl[n]) + id_emb[(size_t)m * DID + n];
    out[(size_t)m * DID + n] = (OutT)lrelu(accG[v] + bg[n] + xhat);
  }
}

extern "C" void kernel_launch(void* const* d_in, const int* in_sizes, int n_in,
                              void* d_out, int out_size, void* d_ws, size_t ws_size,
                              hipStream_t stream) {
  const float* features = (const float*)d_in[0];
  const float* id_emb   = (const float*)d_in[1];
  const float* W_mlp    = (const float*)d_in[2];
  const float* b_mlp    = (const float*)d_in[3];
  const float* W_conv1  = (const float*)d_in[4];
  const float* W_lin1   = (const float*)d_in[5];
  const float* b_lin1   = (const float*)d_in[6];
  const float* W_g1     = (const float*)d_in[7];
  const float* b_g1     = (const float*)d_in[8];
  const float* W_conv2  = (const float*)d_in[9];
  const float* W_lin2   = (const float*)d_in[10];
  const float* b_lin2   = (const float*)d_in[11];
  const float* W_g2     = (const float*)d_in[12];
  const float* b_g2     = (const float*)d_in[13];
  const int*   edge     = (const int*)d_in[14];
  const int E = in_sizes[14] / 2;
  float* out = (float*)d_out;

  // ---- workspace carve-out (all offsets multiples of 2KB) ----
  float*  agg  = (float*)d_ws;                          // 50000*256 f32 (agg1/agg2)
  float*  xw   = agg + (size_t)NNODES * DL;             // 50000*256 f32 (xw1/xw2)
  __bf16* x0b  = (__bf16*)(xw + (size_t)NNODES * DL);   // 50000*256 bf16
  __bf16* x1b  = x0b + (size_t)NNODES * DL;             // 50000*64  bf16
  __bf16* Wmlp = x1b + (size_t)NNODES * DID;            // 256*4096  bf16
  __bf16* Wc1  = Wmlp + (size_t)DL * DF;                // 256*256
  __bf16* Wg1  = Wc1 + DL * DL;                         // 64*256
  __bf16* Wl1  = Wg1 + DID * DL;                        // 64*256
  __bf16* Wc2  = Wl1 + DID * DL;                        // 64*64
  __bf16* Wg2  = Wc2 + DID * DID;                       // 64*64
  __bf16* Wl2  = Wg2 + DID * DID;                       // 64*64

  // ---- pre-convert weights to bf16 (reused many times; kills cvt in loops) --
  k_f32_to_bf16<<<(DL * DF + 255) / 256, 256, 0, stream>>>(W_mlp, Wmlp, DL * DF);
  k_f32_to_bf16<<<(DL * DL + 255) / 256, 256, 0, stream>>>(W_conv1, Wc1, DL * DL);
  k_f32_to_bf16<<<(DID * DL + 255) / 256, 256, 0, stream>>>(W_g1, Wg1, DID * DL);
  k_f32_to_bf16<<<(DID * DL + 255) / 256, 256, 0, stream>>>(W_lin1, Wl1, DID * DL);
  k_f32_to_bf16<<<(DID * DID + 255) / 256, 256, 0, stream>>>(W_conv2, Wc2, DID * DID);
  k_f32_to_bf16<<<(DID * DID + 255) / 256, 256, 0, stream>>>(W_g2, Wg2, DID * DID);
  k_f32_to_bf16<<<(DID * DID + 255) / 256, 256, 0, stream>>>(W_lin2, Wl2, DID * DID);

  const int MB = NNODES / 16;   // 3125
  const int EB = (E + 7) / 8;   // 8 edges (waves) per 256-thread block

  // Stage 1: x0 = l2norm(features @ W_mlp^T + b_mlp)   (bf16 out)
  k_mlp_norm<<<MB, 256, 0, stream>>>(features, Wmlp, b_mlp, x0b);
  // Layer 1 conv projection: xw1 = x0 @ W_conv1^T      (f32 out for atomics)
  k_gemm_nt<DL, DL, 2><<<MB, 256, 0, stream>>>(x0b, Wc1, xw);
  // agg1 = segment_sum(xw1[src] -> dst)
  hipMemsetAsync(agg, 0, (size_t)NNODES * DL * sizeof(float), stream);
  k_scatter<<<EB, 256, 0, stream>>>(xw, edge, E, DL, agg);
  // x1 = lrelu(lrelu(agg1)@W_g1^T + b_g1 + lrelu(x0@W_lin1^T + b_lin1) + id)
  k_combine<DL, __bf16><<<MB, 128, 0, stream>>>(agg, x0b, Wg1, b_g1, Wl1, b_lin1,
                                                id_emb, x1b);
  // Layer 2 conv projection: xw2 = x1 @ W_conv2^T
  k_gemm_nt<DID, DID, 1><<<MB, 128, 0, stream>>>(x1b, Wc2, xw);
  // agg2 = segment_sum(xw2[src] -> dst)
  hipMemsetAsync(agg, 0, (size_t)NNODES * DID * sizeof(float), stream);
  k_scatter<<<EB, 256, 0, stream>>>(xw, edge, E, DID, agg);
  // out = lrelu(lrelu(agg2)@W_g2^T + b_g2 + lrelu(x1@W_lin2^T + b_lin2) + id)
  k_combine<DID, float><<<MB, 128, 0, stream>>>(agg, x1b, Wg2, b_g2, Wl2, b_lin2,
                                                id_emb, out);
}
